// Pedal_15839839388069
// MI455X (gfx1250) — compile-verified, matrix operand order checked by reading, not run on hardware
//
#include <hip/hip_runtime.h>
#include <hip/hip_bf16.h>

// Problem constants (match reference)
#define P_    6
#define M_    512
#define N_    4096
#define D_    1024
#define SCALE_F 15.0f

typedef __attribute__((ext_vector_type(16))) __bf16 v16bf;
typedef __attribute__((ext_vector_type(8)))  __bf16 v8bf;
typedef __attribute__((ext_vector_type(8)))  float  v8f;

// ---------------------------------------------------------------------------
// float -> bf16 with round-to-nearest-even
__device__ __forceinline__ unsigned short f32_to_bf16(float f) {
    unsigned int u = __float_as_uint(f);
    u += 0x7FFFu + ((u >> 16) & 1u);
    return (unsigned short)(u >> 16);
}

// Assemble a per-lane v16bf WMMA operand from two contiguous 8-element chunks
__device__ __forceinline__ v16bf make16(const unsigned short* base, int kA, int kB) {
    v8bf lo = *reinterpret_cast<const v8bf*>(base + kA);
    v8bf hi = *reinterpret_cast<const v8bf*>(base + kB);
    v16bf r;
#pragma unroll
    for (int i = 0; i < 8; ++i) { r[i] = lo[i]; r[i + 8] = hi[i]; }
    return r;
}

__device__ __forceinline__ v8f wmma_bf16(v16bf a, v16bf b, v8f c) {
    return __builtin_amdgcn_wmma_f32_16x16x32_bf16(
        false, a, false, b, (short)0, c, false, false);
}

// ---------------------------------------------------------------------------
// Kernel 0: zero the scalar output (graph-replay safe)
__global__ void zero_out_kernel(float* out) {
    if (threadIdx.x == 0 && blockIdx.x == 0) out[0] = 0.0f;
}

// ---------------------------------------------------------------------------
// Kernel 1: per-row fp32 -> bf16 conversion + squared-norm. One block per row.
__global__ __launch_bounds__(256) void prep_rows_kernel(
    const float* __restrict__ src, unsigned short* __restrict__ dstbf,
    float* __restrict__ sumsq) {
    const int row = blockIdx.x;
    const float* s = src + (size_t)row * D_;
    unsigned short* d = dstbf + (size_t)row * D_;

    float acc = 0.0f;
#pragma unroll
    for (int it = 0; it < D_ / 256; ++it) {
        int i = threadIdx.x + it * 256;
        float v = s[i];
        acc += v * v;
        d[i] = f32_to_bf16(v);
    }
    __shared__ float red[256];
    red[threadIdx.x] = acc;
    __syncthreads();
    for (int s2 = 128; s2 > 0; s2 >>= 1) {
        if (threadIdx.x < (unsigned)s2) red[threadIdx.x] += red[threadIdx.x + s2];
        __syncthreads();
    }
    if (threadIdx.x == 0) sumsq[row] = red[0];
}

// ---------------------------------------------------------------------------
// Kernel 2: batched dist GEMM via bf16 WMMA, software-pipelined.
// Grid = P * (M/16) * (N/512); block = 256 threads (8 waves).
// Each wave: 16x64 tile (4 accumulators) over K=1024. Double-buffered
// operand registers so iteration i+1's 10 b128 loads are in flight while
// iteration i's 4 WMMAs execute (loads complete in order -> no stall).
__global__ __launch_bounds__(256) void gemm_wmma_kernel(
    const unsigned short* __restrict__ fbf, const unsigned short* __restrict__ cbf,
    const float* __restrict__ f2, const float* __restrict__ c2,
    float* __restrict__ dist) {
    const int NCH = N_ / 512;                 // 8 column chunks
    const int bid = blockIdx.x;
    const int nchunk = bid % NCH;
    const int mtile  = (bid / NCH) % (M_ / 16);
    const int p      = bid / (NCH * (M_ / 16));

    const int wave = threadIdx.x >> 5;        // 0..7
    const int lane = threadIdx.x & 31;
    const int half = lane >> 4;               // 0 or 1 (K-half select per ISA layout)
    const int lrow = lane & 15;               // matrix row / column within tile

    const int m0 = mtile * 16;
    const int n0 = nchunk * 512 + wave * 64;  // this wave's 64 columns

    // Per-lane base pointers: A lane holds feature row (m0+lrow);
    // B lane holds centers row (n+lrow) == column of B (K-major in lane).
    const unsigned short* fRow  = fbf + (size_t)(p * M_ + m0 + lrow) * D_;
    const unsigned short* cRow0 = cbf + (size_t)(p * N_ + n0 +  0 + lrow) * D_;
    const unsigned short* cRow1 = cbf + (size_t)(p * N_ + n0 + 16 + lrow) * D_;
    const unsigned short* cRow2 = cbf + (size_t)(p * N_ + n0 + 32 + lrow) * D_;
    const unsigned short* cRow3 = cbf + (size_t)(p * N_ + n0 + 48 + lrow) * D_;

    const int koff = half * 8;                // K-half split across lane halves

    v8f acc0 = {}; v8f acc1 = {}; v8f acc2 = {}; v8f acc3 = {};

    // Prologue: load k-step 0
    v16bf a_c  = make16(fRow,  koff, 16 + koff);
    v16bf b0_c = make16(cRow0, koff, 16 + koff);
    v16bf b1_c = make16(cRow1, koff, 16 + koff);
    v16bf b2_c = make16(cRow2, koff, 16 + koff);
    v16bf b3_c = make16(cRow3, koff, 16 + koff);

#pragma unroll 2
    for (int k0 = 32; k0 < D_; k0 += 32) {
        // Issue next k-step's loads before consuming the current registers.
        v16bf a_n  = make16(fRow,  k0 + koff, k0 + 16 + koff);
        v16bf b0_n = make16(cRow0, k0 + koff, k0 + 16 + koff);
        v16bf b1_n = make16(cRow1, k0 + koff, k0 + 16 + koff);
        v16bf b2_n = make16(cRow2, k0 + koff, k0 + 16 + koff);
        v16bf b3_n = make16(cRow3, k0 + koff, k0 + 16 + koff);

        acc0 = wmma_bf16(a_c, b0_c, acc0);
        acc1 = wmma_bf16(a_c, b1_c, acc1);
        acc2 = wmma_bf16(a_c, b2_c, acc2);
        acc3 = wmma_bf16(a_c, b3_c, acc3);

        a_c = a_n; b0_c = b0_n; b1_c = b1_n; b2_c = b2_n; b3_c = b3_n;
    }

    // Epilogue: final k-step
    acc0 = wmma_bf16(a_c, b0_c, acc0);
    acc1 = wmma_bf16(a_c, b1_c, acc1);
    acc2 = wmma_bf16(a_c, b2_c, acc2);
    acc3 = wmma_bf16(a_c, b3_c, acc3);

    // C/D layout: VGPR r, lane l -> row m0 + r + 8*(l>=16), col ntile + (l%16)
    const int col0 = n0 +  0 + lrow;
    const int col1 = n0 + 16 + lrow;
    const int col2 = n0 + 32 + lrow;
    const int col3 = n0 + 48 + lrow;
    const float c2v0 = c2[p * N_ + col0];
    const float c2v1 = c2[p * N_ + col1];
    const float c2v2 = c2[p * N_ + col2];
    const float c2v3 = c2[p * N_ + col3];
#pragma unroll
    for (int r = 0; r < 8; ++r) {
        const int mrow = m0 + r + 8 * half;
        const float fr = f2[p * M_ + mrow];
        float* drow = dist + (size_t)(p * M_ + mrow) * N_;
        drow[col0] = fr + c2v0 - 2.0f * acc0[r];
        drow[col1] = fr + c2v1 - 2.0f * acc1[r];
        drow[col2] = fr + c2v2 - 2.0f * acc2[r];
        drow[col3] = fr + c2v3 - 2.0f * acc3[r];
    }
}

// ---------------------------------------------------------------------------
// Block-wide reductions over 256 threads (uniform control flow)
__device__ __forceinline__ float block_min256(float v) {
    __shared__ float red[256];
    red[threadIdx.x] = v;
    __syncthreads();
    for (int s = 128; s > 0; s >>= 1) {
        if (threadIdx.x < (unsigned)s)
            red[threadIdx.x] = fminf(red[threadIdx.x], red[threadIdx.x + s]);
        __syncthreads();
    }
    float r = red[0];
    __syncthreads();
    return r;
}

__device__ __forceinline__ float block_sum256(float v) {
    __shared__ float red[256];
    red[threadIdx.x] = v;
    __syncthreads();
    for (int s = 128; s > 0; s >>= 1) {
        if (threadIdx.x < (unsigned)s)
            red[threadIdx.x] += red[threadIdx.x + s];
        __syncthreads();
    }
    float r = red[0];
    __syncthreads();
    return r;
}

// ---------------------------------------------------------------------------
// Kernel 3: per-row selection + logsumexp + hinge. One block per (p, m) row.
__global__ __launch_bounds__(256) void select_rows_kernel(
    const float* __restrict__ dist, const int* __restrict__ camid,
    const int* __restrict__ cam, float* __restrict__ out) {
    const int row = blockIdx.x;               // p*M + m
    const int m = row % M_;
    const float* dr = dist + (size_t)row * N_;
    const int myCam = camid[m];

    float vals[16];
    float wexp[16];
    int   intra[16];
#pragma unroll
    for (int i = 0; i < 16; ++i) {
        const int n = threadIdx.x + i * 256;
        const float dv = dr[n];
        vals[i]  = dv;
        wexp[i]  = __expf(-SCALE_F * dv);
        intra[i] = (cam[n] == myCam) ? 1 : 0;
    }

    // --- extract 6 smallest intra distances (strictly increasing) ---
    float eI[6];
    float lastI = -3.0e38f;
    for (int it = 0; it < 6; ++it) {
        float lm = 3.0e38f;
#pragma unroll
        for (int i = 0; i < 16; ++i)
            if (intra[i] && vals[i] > lastI) lm = fminf(lm, vals[i]);
        lastI = block_min256(lm);
        eI[it] = __expf(-SCALE_F * lastI);
    }

    // --- extract 12 smallest inter distances ---
    float eJ[12];
    float lastJ = -3.0e38f;
    for (int it = 0; it < 12; ++it) {
        float lm = 3.0e38f;
#pragma unroll
        for (int i = 0; i < 16; ++i)
            if (!intra[i] && vals[i] > lastJ) lm = fminf(lm, vals[i]);
        lastJ = block_min256(lm);
        eJ[it] = __expf(-SCALE_F * lastJ);
    }

    // --- exact tail sums beyond the extracted thresholds ---
    float tI = 0.0f, tJ = 0.0f;
#pragma unroll
    for (int i = 0; i < 16; ++i) {
        if (intra[i]  && vals[i] > lastI) tI += wexp[i];
        if (!intra[i] && vals[i] > lastJ) tJ += wexp[i];
    }
    const float tailIntra = block_sum256(tI);
    const float tailInter = block_sum256(tJ);

    if (threadIdx.x == 0) {
        const float x_intra = __logf(fmaxf(eI[0] + eI[1] + eI[2] + eI[3], 1e-37f));
        const float y_intra = __logf(fmaxf(tailIntra, 1e-37f));
        float s8 = 0.0f;
#pragma unroll
        for (int i = 0; i < 8; ++i) s8 += eJ[i];
        const float x_inter = __logf(fmaxf(s8, 1e-37f));
        const float y_inter = __logf(fmaxf(tailInter, 1e-37f));

        const float h = fmaxf(-x_intra + y_intra + 12.0f, 0.0f) +
                        fmaxf(-x_inter + y_inter + 5.0f, 0.0f);
        atomicAdd(out, h * (1.0f / (float)(M_ * P_)));
    }
}

// ---------------------------------------------------------------------------
extern "C" void kernel_launch(void* const* d_in, const int* in_sizes, int n_in,
                              void* d_out, int out_size, void* d_ws, size_t ws_size,
                              hipStream_t stream) {
    (void)in_sizes; (void)n_in; (void)out_size; (void)ws_size;
    const float* feature = (const float*)d_in[0];   // [P, M, D] f32
    const int*   camid   = (const int*)d_in[1];     // [M] i32
    const float* centers = (const float*)d_in[2];   // [P, N, D] f32
    const int*   cam     = (const int*)d_in[3];     // [N] i32
    float* out = (float*)d_out;

    // Workspace layout (all offsets 16B-aligned)
    char* ws = (char*)d_ws;
    size_t off = 0;
    unsigned short* fbf = (unsigned short*)(ws + off); off += (size_t)P_ * M_ * D_ * 2;  // 6.3 MB
    unsigned short* cbf = (unsigned short*)(ws + off); off += (size_t)P_ * N_ * D_ * 2;  // 50.3 MB
    float* f2   = (float*)(ws + off);                  off += (size_t)P_ * M_ * 4;       // 12 KB
    float* c2   = (float*)(ws + off);                  off += (size_t)P_ * N_ * 4;       // 96 KB
    float* dist = (float*)(ws + off);                  // 50.3 MB

    zero_out_kernel<<<1, 64, 0, stream>>>(out);
    prep_rows_kernel<<<P_ * M_, 256, 0, stream>>>(feature, fbf, f2);
    prep_rows_kernel<<<P_ * N_, 256, 0, stream>>>(centers, cbf, c2);
    gemm_wmma_kernel<<<P_ * (M_ / 16) * (N_ / 512), 256, 0, stream>>>(fbf, cbf, f2, c2, dist);
    select_rows_kernel<<<P_ * M_, 256, 0, stream>>>(dist, camid, cam, out);
}
